// MultiHeadedSelfAttention_59536836657437
// MI455X (gfx1250) — matrix operand order repaired and checked by reading, not compiled, not run on hardware
//
#include <hip/hip_runtime.h>

#define N_  2
#define S_  2048
#define D_  1024
#define H_  16
#define DH_ 64

typedef _Float16 f16;
typedef __attribute__((ext_vector_type(16))) _Float16     v16h;
typedef __attribute__((ext_vector_type(8)))  _Float16     v8h;
typedef __attribute__((ext_vector_type(8)))  float        v8f;
typedef __attribute__((ext_vector_type(4)))  unsigned int u32x4;
typedef __attribute__((ext_vector_type(8)))  int          i32x8;
typedef __attribute__((ext_vector_type(4)))  int          i32x4;

union Frag { v16h v; v8h h[2]; };

__device__ __forceinline__ v8f wmma_f16(const Frag& a, const Frag& b, v8f c) {
  // D = A(16x32 f16) * B(32x16 f16) + C(16x16 f32)
  return __builtin_amdgcn_wmma_f32_16x16x32_f16(false, a.v, false, b.v,
                                                (short)0, c, false, false);
}

// ---------------------------------------------------------------------------
// TDM: DMA a 64x64 f16 tile (row-major, row pitch = stride_elems) from global
// memory into LDS at lds_byte_off.  D# layout per CDNA5 ISA ch.8.
// amdgpu-toolchain (clang-23) 6-arg builtin form.
// ---------------------------------------------------------------------------
__device__ __forceinline__ void tdm_load_tile64x64(unsigned int lds_byte_off,
                                                   const f16* gptr,
                                                   unsigned int stride_elems) {
  const unsigned long long ga = (unsigned long long)(uintptr_t)gptr;
  u32x4 g0;
  g0[0] = 1u;                                          // count=1, user mode
  g0[1] = lds_byte_off;                                // lds_addr  (bits 63:32)
  g0[2] = (unsigned int)(ga & 0xffffffffu);            // global_addr[31:0]
  g0[3] = (unsigned int)((ga >> 32) & 0x01ffffffu)     // global_addr[56:32]
        | (2u << 30);                                  // type = 2 ("image")
  i32x8 g1;
  g1[0] = (int)(1u << 16);                             // data_size=1 -> 2 bytes
  g1[1] = (int)(64u << 16);                            // tensor_dim0 = 64
  g1[2] = (int)(64u << 16);                            // (dim0 hi=0) dim1 = 64
  g1[3] = (int)(64u << 16);                            // (dim1 hi=0) tile_dim0=64
  g1[4] = (int)(64u);                                  // tile_dim1=64, tile_dim2=0
  g1[5] = (int)stride_elems;                           // tensor_dim0_stride lo
  g1[6] = 0;                                           // stride hi / dim1_stride lo
  g1[7] = 0;                                           // dim1_stride hi
  const i32x4 z4 = {0, 0, 0, 0};                       // groups 2/3 unused (2-D)
  const i32x8 z8 = {0, 0, 0, 0, 0, 0, 0, 0};
  __builtin_amdgcn_tensor_load_to_lds(g0, g1, z4, z4, z8, 0);
}

__device__ __forceinline__ unsigned int lds_off(const void* p) {
  // generic LDS pointers carry the LDS byte offset in the low 32 bits
  return (unsigned int)(uintptr_t)p;
}

// ---------------------------------------------------------------------------
// Kernel 1: per-head QKV projection.  One wave = one 16-row s-tile of one
// (n,h).  Emits f16 Q (pre-scaled by 1/sqrt(Dh)), f16 K [N,H,S,Dh], and f16
// V transposed [N,H,Dh,S] so attention B-fragments are contiguous.
// ---------------------------------------------------------------------------
__global__ void __launch_bounds__(128)
qkv_proj_kernel(const float* __restrict__ x,
                const float* __restrict__ Wq, const float* __restrict__ bq,
                const float* __restrict__ Wk, const float* __restrict__ bk,
                const float* __restrict__ Wv, const float* __restrict__ bv,
                f16* __restrict__ Qh, f16* __restrict__ Kh, f16* __restrict__ Vt)
{
  const int lane = threadIdx.x & 31;
  const int wave = threadIdx.x >> 5;
  const int st   = blockIdx.x * 4 + wave;        // 0 .. N*H*(S/16)-1
  const int n    = st >> 11;
  const int h    = (st >> 7) & (H_ - 1);
  const int s0   = (st & 127) << 4;
  const int m    = lane & 15;
  const int hlf  = lane >> 4;

  // A fragments of x tile [16 x 64] (f32 -> f16), contiguous k = 16*hlf + 0..15 per lane half
  const float* xb = x + ((size_t)n * S_ + s0) * D_ + h * DH_;
  Frag xa[2];
#pragma unroll
  for (int g = 0; g < 2; ++g) {
    const float* p = xb + (size_t)m * D_ + 32 * g + 16 * hlf;
#pragma unroll
    for (int i = 0; i < 8; ++i) {
      xa[g].v[i]     = (f16)p[i];
      xa[g].v[8 + i] = (f16)p[8 + i];
    }
  }

  const size_t nh = (size_t)n * H_ + h;
  f16* qo = Qh + (nh * S_ + s0) * DH_;
  f16* ko = Kh + (nh * S_ + s0) * DH_;
  f16* vo = Vt + nh * (size_t)DH_ * S_;          // [Dh][S]

  const float* Ws[3] = {Wq, Wk, Wv};
  const float* Bs[3] = {bq, bk, bv};

#pragma unroll
  for (int mtx = 0; mtx < 3; ++mtx) {
    const float* W = Ws[mtx] + (size_t)h * DH_ * DH_;
    const float* B = Bs[mtx] + (size_t)h * DH_;
#pragma unroll
    for (int t = 0; t < 4; ++t) {
      const float bias = B[16 * t + m];
      v8f c;
#pragma unroll
      for (int r = 0; r < 8; ++r) c[r] = bias;
#pragma unroll
      for (int g = 0; g < 2; ++g) {
        // B-frag: col e = 16t+m, rows k = 32g+16*hlf+0..15; B[d][e]=W[e][d]
        Frag wb;
        const float* wp = W + (size_t)(16 * t + m) * DH_ + 32 * g + 16 * hlf;
#pragma unroll
        for (int i = 0; i < 16; ++i) wb.v[i] = (f16)wp[i];
        c = wmma_f16(xa[g], wb, c);
      }
#pragma unroll
      for (int r = 0; r < 8; ++r) {
        const int row = r + 8 * hlf;
        const int col = 16 * t + m;
        if (mtx == 0)      qo[(size_t)row * DH_ + col] = (f16)(0.125f * c[r]);
        else if (mtx == 1) ko[(size_t)row * DH_ + col] = (f16)c[r];
        else               vo[(size_t)col * S_ + (s0 + row)] = (f16)c[r];
      }
    }
  }
}

// ---------------------------------------------------------------------------
// Kernel 2: flash attention.  4 waves/block share one (n,h); each wave owns a
// 16-row Q tile.  K/V chunks (64 keys) are DMA'd into double-buffered LDS by
// the Tensor Data Mover and consumed as WMMA fragments via ds_load_b128.
// ---------------------------------------------------------------------------
__global__ void __launch_bounds__(128)
flash_attn_kernel(const f16* __restrict__ Qh, const f16* __restrict__ Kh,
                  const f16* __restrict__ Vt, float* __restrict__ out)
{
  __shared__ f16 Kls[2][64 * 64];                // 2 x 8KB : keys x dh
  __shared__ f16 Vls[2][64 * 64];                // 2 x 8KB : dh  x keys
  __shared__ f16 pbuf[4][16 * 64];               // per-wave P staging, 2KB ea

  const int lane = threadIdx.x & 31;
  const int wave = threadIdx.x >> 5;
  const int qt   = blockIdx.x * 4 + wave;        // 0..4095
  const int n    = qt >> 11;
  const int h    = (qt >> 7) & (H_ - 1);
  const int q0   = (qt & 127) << 4;
  const int m    = lane & 15;
  const int hlf  = lane >> 4;

  const size_t nh = (size_t)n * H_ + h;
  const f16* qb = Qh + (nh * S_ + q0) * DH_;
  const f16* kb = Kh + nh * (size_t)S_ * DH_;    // [S][Dh]
  const f16* vb = Vt + nh * (size_t)DH_ * S_;    // [Dh][S]
  f16* pb = pbuf[wave];

  // Q A-fragments (already scaled by 1/sqrt(Dh))
  Frag qa[2];
#pragma unroll
  for (int g = 0; g < 2; ++g) {
    const f16* p = qb + (size_t)m * DH_ + 32 * g + 16 * hlf;
    qa[g].h[0] = *(const v8h*)p;
    qa[g].h[1] = *(const v8h*)(p + 8);
  }

  v8f o[4];
  float mrun[8], lrun[8];
#pragma unroll
  for (int t = 0; t < 4; ++t)
#pragma unroll
    for (int r = 0; r < 8; ++r) o[t][r] = 0.f;
#pragma unroll
  for (int r = 0; r < 8; ++r) { mrun[r] = -3.0e38f; lrun[r] = 0.f; }

  constexpr int NC = S_ / 64;                    // 32 key chunks

  // prologue: DMA chunk 0 into buffer 0 (wave 0 drives the TDM)
  if (wave == 0) {
    tdm_load_tile64x64(lds_off(&Kls[0][0]), kb, DH_);
    tdm_load_tile64x64(lds_off(&Vls[0][0]), vb, S_);
  }

  for (int c = 0; c < NC; ++c) {
    const int buf = c & 1;
    __syncthreads();                             // prior reads of buf^1 done
    if (wave == 0) {
      if (c + 1 < NC) {
        tdm_load_tile64x64(lds_off(&Kls[buf ^ 1][0]),
                           kb + (size_t)(c + 1) * 64 * DH_, DH_);
        tdm_load_tile64x64(lds_off(&Vls[buf ^ 1][0]),
                           vb + (size_t)(c + 1) * 64, S_);
        __builtin_amdgcn_s_wait_tensorcnt(2);    // chunk c's two DMAs done
      } else {
        __builtin_amdgcn_s_wait_tensorcnt(0);
      }
    }
    __syncthreads();                             // chunk c visible to all waves
    const f16* kls = &Kls[buf][0];
    const f16* vls = &Vls[buf][0];

    // -------- scores: four 16x16 tiles over 64 keys (pre-scaled via Q) -----
    v8f s[4];
#pragma unroll
    for (int ct = 0; ct < 4; ++ct) {
      v8f acc;
#pragma unroll
      for (int r = 0; r < 8; ++r) acc[r] = 0.f;
#pragma unroll
      for (int g = 0; g < 2; ++g) {
        // B-frag: col = local key 16*ct+m, rows k = 32g+16*hlf+0..15 over d
        Frag kf;
        const f16* p = kls + (size_t)(16 * ct + m) * DH_ + 32 * g + 16 * hlf;
        kf.h[0] = *(const v8h*)p;
        kf.h[1] = *(const v8h*)(p + 8);
        acc = wmma_f16(qa[g], kf, acc);
      }
      s[ct] = acc;
    }

    // -------- online softmax (row = r + 8*hlf; cols across 16 lanes) -------
#pragma unroll
    for (int r = 0; r < 8; ++r) {
      float t = fmaxf(fmaxf(s[0][r], s[1][r]), fmaxf(s[2][r], s[3][r]));
#pragma unroll
      for (int msk = 1; msk < 16; msk <<= 1)
        t = fmaxf(t, __shfl_xor(t, msk, 32));
      const float mn    = fmaxf(mrun[r], t);
      const float alpha = __expf(mrun[r] - mn);
      mrun[r] = mn;
      float rs = 0.f;
#pragma unroll
      for (int ct = 0; ct < 4; ++ct) {
        const float pv = __expf(s[ct][r] - mn);
        s[ct][r] = pv;
        rs += pv;
      }
#pragma unroll
      for (int msk = 1; msk < 16; msk <<= 1)
        rs += __shfl_xor(rs, msk, 32);
      lrun[r] = lrun[r] * alpha + rs;
#pragma unroll
      for (int t4 = 0; t4 < 4; ++t4) o[t4][r] *= alpha;
    }

    // -------- transpose P (16x64) C-layout -> A-layout through LDS ---------
#pragma unroll
    for (int ct = 0; ct < 4; ++ct)
#pragma unroll
      for (int r = 0; r < 8; ++r)
        pb[(r + 8 * hlf) * 64 + 16 * ct + m] = (f16)s[ct][r];
    asm volatile("s_wait_dscnt 0x0" ::: "memory");   // wave-local LDS RAW
    Frag pa[2];
#pragma unroll
    for (int u = 0; u < 2; ++u) {
      const f16* p = pb + m * 64 + 32 * u + 16 * hlf;
      pa[u].h[0] = *(const v8h*)p;
      pa[u].h[1] = *(const v8h*)(p + 8);
    }
    asm volatile("" ::: "memory");

    // -------- O += P (16x64) * V (64x64), two k=32 steps -------------------
#pragma unroll
    for (int t4 = 0; t4 < 4; ++t4) {
      v8f acc = o[t4];
#pragma unroll
      for (int u = 0; u < 2; ++u) {
        // B-frag from V LDS tile: col e = 16*t4+m, k = 32u+16*hlf+0..15
        Frag vf;
        const f16* p = vls + (size_t)(16 * t4 + m) * 64 + 32 * u + 16 * hlf;
        vf.h[0] = *(const v8h*)p;
        vf.h[1] = *(const v8h*)(p + 8);
        acc = wmma_f16(pa[u], vf, acc);
      }
      o[t4] = acc;
    }
  }

  // -------- epilogue: divide by row sums, store fp32 -----------------------
  float* ob = out + ((size_t)n * S_ + q0) * D_ + h * DH_;
#pragma unroll
  for (int r = 0; r < 8; ++r) {
    const float inv = 1.0f / lrun[r];
#pragma unroll
    for (int t4 = 0; t4 < 4; ++t4)
      ob[(size_t)(r + 8 * hlf) * D_ + 16 * t4 + m] = o[t4][r] * inv;
  }
}

// ---------------------------------------------------------------------------
extern "C" void kernel_launch(void* const* d_in, const int* in_sizes, int n_in,
                              void* d_out, int out_size, void* d_ws, size_t ws_size,
                              hipStream_t stream) {
  const float* x  = (const float*)d_in[0];
  const float* Wq = (const float*)d_in[1];
  const float* bq = (const float*)d_in[2];
  const float* Wk = (const float*)d_in[3];
  const float* bk = (const float*)d_in[4];
  const float* Wv = (const float*)d_in[5];
  const float* bv = (const float*)d_in[6];
  float* out = (float*)d_out;

  const size_t per = (size_t)N_ * H_ * S_ * DH_;   // 4,194,304 f16 elems
  f16* Qh = (f16*)d_ws;
  f16* Kh = Qh + per;
  f16* Vt = Kh + per;                              // total 24 MB of ws

  const int tiles  = N_ * H_ * (S_ / 16);          // 4096 wave-tiles
  const int blocks = tiles / 4;                    // 4 waves (128 thr) / block

  qkv_proj_kernel<<<dim3(blocks), dim3(128), 0, stream>>>(
      x, Wq, bq, Wk, bk, Wv, bv, Qh, Kh, Vt);
  flash_attn_kernel<<<dim3(blocks), dim3(128), 0, stream>>>(
      Qh, Kh, Vt, out);
}
